// MultiHeadAttention_19344532701482
// MI455X (gfx1250) — matrix operand order repaired and checked by reading, not compiled, not run on hardware
//
#include <hip/hip_runtime.h>

// MHA forward on gfx1250 (MI455X): bf16 WMMA (16x16x32), f32 accumulate.
// B=2 S=2048 D=1024 H=16 DH=64. ~69 GFLOP; all tensors fit in 192MB L2 ->
// matrix-core bound. bf16 WMMA gives 8x K-depth per issue vs f32 WMMA.
// V is stored transposed [B,H,DH,S] by the QKV kernel so attention needs no
// LDS staging / barriers; Q is pre-scaled by 1/sqrt(D).
// GEMM B-operands are streamed through an explicit 3-slot fragment ring so
// each WMMA runs with the next fragments' loads already in flight.

constexpr int NB = 2;
constexpr int NS = 2048;
constexpr int ND = 1024;
constexpr int NH = 16;
constexpr int NDH = 64;

typedef __attribute__((ext_vector_type(16))) __bf16 v16bf;
typedef __attribute__((ext_vector_type(8)))  __bf16 bf16x8;
typedef __attribute__((ext_vector_type(8)))  float  v8f;
typedef __attribute__((ext_vector_type(4)))  float  f32x4;

union Frag16 { v16bf v; bf16x8 h[2]; };

static __device__ __forceinline__ __bf16 f2bf(float x) { return (__bf16)x; }

static __device__ __forceinline__ v8f vzero8() {
  v8f z;
#pragma unroll
  for (int i = 0; i < 8; ++i) z[i] = 0.0f;
  return z;
}

static __device__ __forceinline__ v8f wmma_bf16(v16bf a, v16bf b, v8f c) {
  // (neg_a, A, neg_b, B, c_mod, C, reuse_a, reuse_b)
  return __builtin_amdgcn_wmma_f32_16x16x32_bf16(false, a, false, b, (short)0, c,
                                                 false, false);
}

// B fragment / contiguous 16 bf16 (two 16B chunks at +0,+8 elements).
static __device__ __forceinline__ void load_b16(Frag16& f, const __bf16* p) {
  f.h[0] = *(const bf16x8*)(p);
  f.h[1] = *(const bf16x8*)(p + 8);
}
// A fragment from bf16 row: contiguous 16 K (halves at K=+0 and K=+8).
static __device__ __forceinline__ void load_a16(Frag16& f, const __bf16* p) {
  f.h[0] = *(const bf16x8*)(p);
  f.h[1] = *(const bf16x8*)(p + 8);
}
// A fragment from f32 row with on-the-fly bf16 convert.
static __device__ __forceinline__ void load_a32(Frag16& f, const float* p) {
  f32x4 x0 = *(const f32x4*)(p);
  f32x4 x1 = *(const f32x4*)(p + 4);
  f32x4 x2 = *(const f32x4*)(p + 8);
  f32x4 x3 = *(const f32x4*)(p + 12);
#pragma unroll
  for (int i = 0; i < 4; ++i) {
    f.v[i] = f2bf(x0[i]);
    f.v[4 + i] = f2bf(x1[i]);
    f.v[8 + i] = f2bf(x2[i]);
    f.v[12 + i] = f2bf(x3[i]);
  }
}

// ---------------------------------------------------------------------------
// Prep: W[batch][K][N] (f32) -> WT[batch][N][K] (bf16). Grid covers all elems.
// ---------------------------------------------------------------------------
__global__ __launch_bounds__(256) void k_transpose_bf16(
    const float* __restrict__ in, __bf16* __restrict__ out, int Kdim, int Ndim) {
  int idx = blockIdx.x * 256 + threadIdx.x;
  int per = Kdim * Ndim;
  int b = idx / per;
  int rem = idx - b * per;
  int n = rem / Kdim;
  int k = rem - n * Kdim;
  out[(long long)b * per + (long long)n * Kdim + k] =
      f2bf(in[(long long)b * per + (long long)k * Ndim + n]);
}

// ---------------------------------------------------------------------------
// QKV projection: per (b,h), C[S x 64] = X[b] (S x 1024) * W[h] (1024 x 64).
// Wave computes a 16-row strip of Q, K and V (shares the A fragment).
// 12 WMMAs per k-step; B fragments flow through a 3-slot ring (loads issued
// two fragments ahead of consumption). Next A fragment prefetched each step.
// Q is scaled by 1/sqrt(D)=1/32; V stored transposed [B,H,DH,S].
// grid = (S/128, B*H), block = 256 (8 waves).
// ---------------------------------------------------------------------------
__global__ __launch_bounds__(256) void k_qkv(
    const float* __restrict__ X, const __bf16* __restrict__ WqT,
    const __bf16* __restrict__ WkT, const __bf16* __restrict__ WvT,
    __bf16* __restrict__ Q, __bf16* __restrict__ K, __bf16* __restrict__ VT) {
  const int tid = threadIdx.x;
  const int wv = tid >> 5;
  const int lane = tid & 31;
  const int half = lane >> 4;
  const int l16 = lane & 15;
  const int bh = blockIdx.y;           // b*H + h
  const int h = bh & 15;
  const int r0 = blockIdx.x * 128 + wv * 16;

  v8f acc[3][4];
#pragma unroll
  for (int m = 0; m < 3; ++m)
#pragma unroll
    for (int nt = 0; nt < 4; ++nt) acc[m][nt] = vzero8();

  const float* xrow = X + ((long long)((bh >> 4) * NS + r0 + l16)) * ND;

  // Per-lane weight base pointers; fragment (m,nt,k0) = wbase[m]+nt*16*ND+k0.
  const __bf16* wbase[3];
  wbase[0] = WqT + ((long long)(h * 64 + l16)) * ND + half * 16;
  wbase[1] = WkT + ((long long)(h * 64 + l16)) * ND + half * 16;
  wbase[2] = WvT + ((long long)(h * 64 + l16)) * ND + half * 16;

  Frag16 a_cur, a_nxt;
  load_a32(a_cur, xrow + half * 16);

  // Prime the B-fragment ring with sequence elements 0 and 1 (k0 = 0).
  Frag16 bb[3];
  load_b16(bb[0], wbase[0]);
  load_b16(bb[1], wbase[0] + (long long)16 * ND);

  for (int k0 = 0; k0 < ND; k0 += 32) {
    load_a32(a_nxt, xrow + ((k0 + 32) & (ND - 1)) + half * 16);
    const int k0n = (k0 + 32) & (ND - 1);  // wrap: last-iter loads are dummies
#pragma unroll
    for (int j = 0; j < 12; ++j) {
      const int m = j >> 2, nt = j & 3;
      const int s2 = j + 2;                      // fragment two ahead
      const int jm = (s2 >= 12) ? (s2 - 12) : s2;
      const int kk = (s2 >= 12) ? k0n : k0;      // compile-time select per j
      load_b16(bb[s2 % 3], wbase[jm >> 2] + (long long)(jm & 3) * 16 * ND + kk);
      acc[m][nt] = wmma_bf16(a_cur.v, bb[j % 3].v, acc[m][nt]);
    }
    a_cur = a_nxt;
  }

#pragma unroll
  for (int nt = 0; nt < 4; ++nt)
#pragma unroll
    for (int r = 0; r < 8; ++r) {
      int row = r0 + r + half * 8;            // C layout: lanes16-31 -> M+8
      int col = nt * 16 + l16;
      // Q scaled by 1/sqrt(D); row-major [B,H,S,DH]
      Q[((long long)bh * NS + row) * NDH + col] = f2bf(acc[0][nt][r] * 0.03125f);
      K[((long long)bh * NS + row) * NDH + col] = f2bf(acc[1][nt][r]);
      // V transposed: [B,H,DH,S]
      VT[((long long)bh * NDH + col) * NS + row] = f2bf(acc[2][nt][r]);
    }
}

// ---------------------------------------------------------------------------
// Flash attention per (b,h): wave owns 16 query rows, streams 32 keys/iter.
// K fragments pipelined one block ahead; V read straight from transposed
// global layout (B fragments here use 16 contiguous K per lane). No barriers.
// grid = (S/128, B*H), block = 256 (8 waves).
// ---------------------------------------------------------------------------
__global__ __launch_bounds__(256) void k_attn(
    const __bf16* __restrict__ Q, const __bf16* __restrict__ K,
    const __bf16* __restrict__ VT, __bf16* __restrict__ O) {
  __shared__ __bf16 sPT[8 * 16 * 40];      // per-wave P tile: [row][col], pad 40

  const int tid = threadIdx.x;
  const int wv = tid >> 5;
  const int lane = tid & 31;
  const int half = lane >> 4;
  const int l16 = lane & 15;
  const int bh = blockIdx.y;
  const int b = bh >> 4, h = bh & 15;
  const int r0 = blockIdx.x * 128 + wv * 16;
  const long long base = (long long)bh * NS * NDH;   // Q/K base
  const __bf16* vbase = VT + base;                   // [DH][S] panel

  // Q fragments (pre-scaled by 1/sqrt(D)), resident for whole kernel.
  Frag16 qf[2];
#pragma unroll
  for (int kk = 0; kk < 2; ++kk)
    load_a16(qf[kk], Q + base + (long long)(r0 + l16) * NDH + kk * 32 + half * 16);

  float mrow[8], lrow[8];
  v8f acc[4];
#pragma unroll
  for (int r = 0; r < 8; ++r) { mrow[r] = -1.0e30f; lrow[r] = 0.0f; }
#pragma unroll
  for (int nt = 0; nt < 4; ++nt) acc[nt] = vzero8();

  __bf16* pt = sPT + wv * 16 * 40;

  // Preload K fragments for first key block. kf[t][kk]: t = 16-col tile.
  Frag16 kf[2][2];
#pragma unroll
  for (int t = 0; t < 2; ++t)
#pragma unroll
    for (int kk = 0; kk < 2; ++kk)
      load_b16(kf[t][kk],
               K + base + (long long)(t * 16 + l16) * NDH + kk * 32 + half * 16);

  for (int kb = 0; kb < NS; kb += 32) {
    // S = Q * K^T for 32 keys (two 16-col tiles). Scale already in Q.
    v8f s0 = vzero8(), s1 = vzero8();
#pragma unroll
    for (int kk = 0; kk < 2; ++kk) {
      s0 = wmma_bf16(qf[kk].v, kf[0][kk].v, s0);
      s1 = wmma_bf16(qf[kk].v, kf[1][kk].v, s1);
    }

    // Pipeline: K fragments for next key block (wraps on last iteration).
    const int kn = (kb + 32) & (NS - 1);
#pragma unroll
    for (int t = 0; t < 2; ++t)
#pragma unroll
      for (int kk = 0; kk < 2; ++kk)
        load_b16(kf[t][kk], K + base + (long long)(kn + t * 16 + l16) * NDH +
                                kk * 32 + half * 16);

    // V fragments for this block: VT[n][kb + half*16 + 0..15], contiguous.
    Frag16 vf[4];
#pragma unroll
    for (int nt = 0; nt < 4; ++nt)
      load_b16(vf[nt], vbase + (long long)(nt * 16 + l16) * NS + kb + half * 16);

    // Prefetch K/V streams a few blocks ahead (global_prefetch_b8).
    {
      const int kp = (kb + 256) & (NS - 1);
      __builtin_prefetch(K + base + (long long)(kp + l16) * NDH, 0, 1);
      __builtin_prefetch(vbase + (long long)(l16 * 4) * NS + kp, 0, 1);
    }

    // Online softmax; each row lives on one 16-lane group -> shfl_xor 1,2,4,8.
#pragma unroll
    for (int r = 0; r < 8; ++r) {
      float a0 = s0[r], a1 = s1[r];
      float rm = fmaxf(a0, a1);
      rm = fmaxf(rm, __shfl_xor(rm, 1));
      rm = fmaxf(rm, __shfl_xor(rm, 2));
      rm = fmaxf(rm, __shfl_xor(rm, 4));
      rm = fmaxf(rm, __shfl_xor(rm, 8));
      float mn = fmaxf(mrow[r], rm);
      float fac = __expf(mrow[r] - mn);
      float p0 = __expf(a0 - mn);
      float p1 = __expf(a1 - mn);
      float rs = p0 + p1;
      rs += __shfl_xor(rs, 1);
      rs += __shfl_xor(rs, 2);
      rs += __shfl_xor(rs, 4);
      rs += __shfl_xor(rs, 8);
      lrow[r] = lrow[r] * fac + rs;
      mrow[r] = mn;
#pragma unroll
      for (int nt = 0; nt < 4; ++nt) acc[nt][r] *= fac;
      int prow = r + half * 8;                     // C-layout row of this lane
      pt[prow * 40 + l16] = f2bf(p0);
      pt[prow * 40 + 16 + l16] = f2bf(p1);
    }
    // LDS round-trip rearranges P from C-layout to A-layout (same wave only;
    // DS ops are in-order within a wave, just fence the counter + compiler).
    asm volatile("s_wait_dscnt 0" ::: "memory");
    Frag16 pf;
    load_a16(pf, pt + l16 * 40 + half * 16);

    // O += P (16x32) * V (32x64)
#pragma unroll
    for (int nt = 0; nt < 4; ++nt)
      acc[nt] = wmma_bf16(pf.v, vf[nt].v, acc[nt]);
  }

  // Normalize and store heads concatenated: O[b, s, h*64+col] (bf16).
#pragma unroll
  for (int r = 0; r < 8; ++r) {
    float inv = 1.0f / lrow[r];
    int row = r0 + r + half * 8;
#pragma unroll
    for (int nt = 0; nt < 4; ++nt) {
      int col = h * 64 + nt * 16 + l16;
      O[((long long)(b * NS + row)) * ND + col] = f2bf(acc[nt][r] * inv);
    }
  }
}

// ---------------------------------------------------------------------------
// Output projection: Y = O (4096 x 1024, bf16) * Wo (1024 x 1024) + bo, f32.
// 3-slot B-fragment ring (loads two fragments ahead) + next-A prefetch.
// grid = (B*S/128, D/64), block = 256 (8 waves).
// ---------------------------------------------------------------------------
__global__ __launch_bounds__(256) void k_proj(
    const __bf16* __restrict__ O, const __bf16* __restrict__ WoT,
    const float* __restrict__ bo, float* __restrict__ Y) {
  const int tid = threadIdx.x;
  const int wv = tid >> 5;
  const int lane = tid & 31;
  const int half = lane >> 4;
  const int l16 = lane & 15;
  const int r0 = blockIdx.x * 128 + wv * 16;
  const int n0 = blockIdx.y * 64;

  v8f acc[4];
#pragma unroll
  for (int nt = 0; nt < 4; ++nt) acc[nt] = vzero8();

  const __bf16* arow = O + (long long)(r0 + l16) * ND;
  const __bf16* wb = WoT + (long long)(n0 + l16) * ND + half * 16;

  Frag16 a_cur, a_nxt;
  load_a16(a_cur, arow + half * 16);

  Frag16 bb[3];
  load_b16(bb[0], wb);
  load_b16(bb[1], wb + (long long)16 * ND);

  for (int k0 = 0; k0 < ND; k0 += 32) {
    load_a16(a_nxt, arow + ((k0 + 32) & (ND - 1)) + half * 16);
    const int k0n = (k0 + 32) & (ND - 1);
#pragma unroll
    for (int nt = 0; nt < 4; ++nt) {
      const int s2 = nt + 2;
      const int jn = s2 & 3;
      const int kk = (s2 >= 4) ? k0n : k0;       // compile-time select per nt
      load_b16(bb[s2 % 3], wb + (long long)jn * 16 * ND + kk);
      acc[nt] = wmma_bf16(a_cur.v, bb[nt % 3].v, acc[nt]);
    }
    a_cur = a_nxt;
  }

#pragma unroll
  for (int nt = 0; nt < 4; ++nt) {
    float bias = bo[n0 + nt * 16 + l16];
#pragma unroll
    for (int r = 0; r < 8; ++r) {
      int row = r0 + r + half * 8;
      Y[(long long)row * ND + n0 + nt * 16 + l16] = acc[nt][r] + bias;
    }
  }
}

// ---------------------------------------------------------------------------
extern "C" void kernel_launch(void* const* d_in, const int* in_sizes, int n_in,
                              void* d_out, int out_size, void* d_ws,
                              size_t ws_size, hipStream_t stream) {
  (void)in_sizes; (void)n_in; (void)out_size; (void)ws_size;
  const float* X  = (const float*)d_in[0];
  const float* Wq = (const float*)d_in[1];
  const float* Wk = (const float*)d_in[2];
  const float* Wv = (const float*)d_in[3];
  const float* Wo = (const float*)d_in[4];
  const float* bo = (const float*)d_in[5];
  float* Y = (float*)d_out;

  char* ws = (char*)d_ws;
  const size_t MB = 1024ull * 1024ull;
  __bf16* Qb  = (__bf16*)(ws + 0 * MB);    // [B,H,S,DH] bf16 : 8 MB (pre-scaled)
  __bf16* Kb  = (__bf16*)(ws + 8 * MB);    // [B,H,S,DH] bf16 : 8 MB
  __bf16* Vt  = (__bf16*)(ws + 16 * MB);   // [B,H,DH,S] bf16 : 8 MB (transposed)
  __bf16* Ob  = (__bf16*)(ws + 24 * MB);   // [B,S,D]    bf16 : 8 MB
  __bf16* WqT = (__bf16*)(ws + 32 * MB);   // [H,DH,D]   bf16 : 2 MB
  __bf16* WkT = (__bf16*)(ws + 34 * MB);
  __bf16* WvT = (__bf16*)(ws + 36 * MB);
  __bf16* WoT = (__bf16*)(ws + 38 * MB);   // [D,D] (N-major) bf16 : 2 MB

  // Weight prep (transpose + f32->bf16). 4M elems total, trivially cheap.
  k_transpose_bf16<<<(NH * ND * NDH) / 256, 256, 0, stream>>>(Wq, WqT, ND, NDH);
  k_transpose_bf16<<<(NH * ND * NDH) / 256, 256, 0, stream>>>(Wk, WkT, ND, NDH);
  k_transpose_bf16<<<(NH * ND * NDH) / 256, 256, 0, stream>>>(Wv, WvT, ND, NDH);
  k_transpose_bf16<<<(ND * ND) / 256, 256, 0, stream>>>(Wo, WoT, ND, ND);

  k_qkv<<<dim3(NS / 128, NB * NH), 256, 0, stream>>>(X, WqT, WkT, WvT, Qb, Kb, Vt);
  k_attn<<<dim3(NS / 128, NB * NH), 256, 0, stream>>>(Qb, Kb, Vt, Ob);
  k_proj<<<dim3((NB * NS) / 128, ND / 64), 256, 0, stream>>>(Ob, WoT, bo, Y);
}
